// mLSTMCell_68453188763733
// MI455X (gfx1250) — compile-verified, matrix operand order checked by reading, not run on hardware
//
#include <hip/hip_runtime.h>
#include <math.h>

// Problem constants (from reference setup_inputs).
#define B_  2
#define S_  2048
#define E_  1024
#define NH_ 8
#define DH_ 128

#define BM 64          // rows per workgroup (16 per wave, 4 waves)
#define BN 32          // j-columns per tile
#define VTS 40         // padded LDS stride (ushorts) for transposed V
#define OST 132        // padded LDS stride (floats) for output staging

typedef __attribute__((ext_vector_type(16))) __bf16 v16bf;
typedef __attribute__((ext_vector_type(8)))  float  v8f;
typedef __attribute__((ext_vector_type(8)))  unsigned short ushort8;
typedef __attribute__((ext_vector_type(4)))  float  f4;

union Frag { v16bf v; ushort8 h2[2]; unsigned short u[16]; };

__device__ __forceinline__ unsigned short f2bf(float f) {
  union { float f; unsigned int u; } x; x.f = f;
  unsigned int r = x.u + 0x7FFFu + ((x.u >> 16) & 1u);  // round-to-nearest-even
  return (unsigned short)(r >> 16);
}
__device__ __forceinline__ float bf2f(unsigned short b) {
  union { unsigned int u; float f; } x; x.u = ((unsigned int)b) << 16;
  return x.f;
}

// ---------------------------------------------------------------------------
// Kernel 1: gate pre-activations  igp/fgp[(b*NH+h)*S + s] = qkv @ W + bias
// ---------------------------------------------------------------------------
__global__ void gates_kernel(const float* __restrict__ q, const float* __restrict__ k,
                             const float* __restrict__ v,
                             const float* __restrict__ igk, const float* __restrict__ igb,
                             const float* __restrict__ fgk, const float* __restrict__ fgb,
                             float* __restrict__ igp, float* __restrict__ fgp) {
  int b = blockIdx.x / S_;
  int s = blockIdx.x % S_;
  int tid = threadIdx.x;
  float ia[NH_], fa[NH_];
  #pragma unroll
  for (int h = 0; h < NH_; h++) { ia[h] = 0.f; fa[h] = 0.f; }
  size_t rb = ((size_t)(b * S_ + s)) * E_;
  for (int e = tid; e < 3 * E_; e += 256) {
    float val = (e < E_) ? q[rb + e] : (e < 2 * E_) ? k[rb + e - E_] : v[rb + e - 2 * E_];
    const float* ir = igk + (size_t)e * NH_;
    const float* fr = fgk + (size_t)e * NH_;
    #pragma unroll
    for (int h = 0; h < NH_; h++) { ia[h] += val * ir[h]; fa[h] += val * fr[h]; }
  }
  __shared__ float red[256];
  for (int h = 0; h < NH_; h++) {
    red[tid] = ia[h]; __syncthreads();
    for (int o = 128; o > 0; o >>= 1) { if (tid < o) red[tid] += red[tid + o]; __syncthreads(); }
    if (tid == 0) igp[((size_t)(b * NH_ + h)) * S_ + s] = red[0] + igb[h];
    __syncthreads();
    red[tid] = fa[h]; __syncthreads();
    for (int o = 128; o > 0; o >>= 1) { if (tid < o) red[tid] += red[tid + o]; __syncthreads(); }
    if (tid == 0) fgp[((size_t)(b * NH_ + h)) * S_ + s] = red[0] + fgb[h];
    __syncthreads();
  }
}

// ---------------------------------------------------------------------------
// Kernel 2: per-(b,h) inclusive scan of log_sigmoid(fg) -> cs;  b_j = ig - cs
// ---------------------------------------------------------------------------
__global__ void scan_kernel(const float* __restrict__ igp, const float* __restrict__ fgp,
                            float* __restrict__ csum, float* __restrict__ bcol) {
  int bh = blockIdx.x; int tid = threadIdx.x;
  size_t base = (size_t)bh * S_;
  const int CH = S_ / 256;  // 8 elements per thread
  float loc[CH]; float run = 0.f;
  #pragma unroll
  for (int c = 0; c < CH; c++) {
    int s = tid * CH + c;
    float x = fgp[base + s];
    float ls = fminf(x, 0.f) - log1pf(__expf(-fabsf(x)));  // log_sigmoid
    run += ls; loc[c] = run;
  }
  __shared__ float part[256];
  part[tid] = run; __syncthreads();
  for (int off = 1; off < 256; off <<= 1) {
    float t = (tid >= off) ? part[tid - off] : 0.f;
    __syncthreads();
    part[tid] += t;
    __syncthreads();
  }
  float excl = part[tid] - run;
  #pragma unroll
  for (int c = 0; c < CH; c++) {
    int s = tid * CH + c;
    float cv = loc[c] + excl;
    csum[base + s] = cv;
    bcol[base + s] = igp[base + s] - cv;
  }
}

// ---------------------------------------------------------------------------
// Kernel 3: stabilized mLSTM attention + multihead layernorm (WMMA bf16)
// ---------------------------------------------------------------------------
__launch_bounds__(128)
__global__ void mlstm_attn_kernel(const float* __restrict__ q, const float* __restrict__ k,
                                  const float* __restrict__ v,
                                  const float* __restrict__ csum, const float* __restrict__ bcol,
                                  const float* __restrict__ lnw, float* __restrict__ out) {
  __shared__ __align__(16) unsigned short Kbuf[BN * DH_];     // K tile, row-major bf16
  __shared__ __align__(16) unsigned short VTb[DH_ * VTS];     // V tile, transposed bf16
  __shared__ __align__(16) unsigned short Pst[4][16 * BN];    // per-wave P staging (bf16)
  __shared__ float bvals[BN];
  __shared__ float mrow[4][16], srow[4][16], scm[4][16], emm[4][16];
  __shared__ float outst[4][16 * OST];

  const int t = threadIdx.x;
  const int w = t >> 5;
  const int lane = t & 31;
  const int g = lane >> 4;     // lane half (K/M split per WMMA layout)
  const int ln = lane & 15;

  const int tiles = S_ / BM;
  int gid = blockIdx.x;
  int it = gid % tiles;
  int bh = gid / tiles;
  int h = bh % NH_;
  int b = bh / NH_;
  const int i0 = it * BM;
  const int irow0 = i0 + w * 16;
  const size_t gbase = (size_t)bh * S_;
  const float invsqrt = 0.08838834764831845f;  // 1/sqrt(DH)

  // Per-row constant a_i = cs[i]; running stats init (lanes 0..15 own rows 0..15).
  float a_l = 0.f;
  if (lane < 16) {
    a_l = csum[gbase + irow0 + lane];
    mrow[w][lane] = -__builtin_inff();
    srow[w][lane] = 0.f;
  }

  // Q A-fragments: 4 K-chunks of 16x32 bf16 per wave, loaded straight from global.
  // A layout: lane half g=0 holds K = kb..kb+7 and kb+16..kb+23; g=1 holds +8.
  Frag qa[4];
  {
    const float* qp = q + ((size_t)(b * S_ + irow0 + ln)) * E_ + h * DH_;
    const int kb0 = g * 8;
    #pragma unroll
    for (int c = 0; c < 4; c++) {
      int kb = c * 32 + kb0;
      f4 f0 = *(const f4*)(qp + kb);
      f4 f1 = *(const f4*)(qp + kb + 4);
      f4 f2 = *(const f4*)(qp + kb + 16);
      f4 f3 = *(const f4*)(qp + kb + 20);
      #pragma unroll
      for (int e = 0; e < 4; e++) {
        qa[c].u[e]      = f2bf(f0[e]);
        qa[c].u[4 + e]  = f2bf(f1[e]);
        qa[c].u[8 + e]  = f2bf(f2[e]);
        qa[c].u[12 + e] = f2bf(f3[e]);
      }
    }
  }

  const v8f vz = {0.f, 0.f, 0.f, 0.f, 0.f, 0.f, 0.f, 0.f};
  v8f acc[8];
  #pragma unroll
  for (int d = 0; d < 8; d++) acc[d] = vz;

  const int njt = i0 / BN + 2;  // causal j-tile range for this 64-row block
  for (int jt = 0; jt < njt; jt++) {
    const int j0 = jt * BN;
    // ---- cooperative tile load: K row-major, V transposed, gate columns ----
    {
      int r = t >> 2, c0 = (t & 3) * 32;
      const float* kp = k + ((size_t)(b * S_ + j0 + r)) * E_ + h * DH_ + c0;
      const float* vp = v + ((size_t)(b * S_ + j0 + r)) * E_ + h * DH_ + c0;
      #pragma unroll
      for (int m = 0; m < 8; m++) {
        f4 x = ((const f4*)kp)[m];
        #pragma unroll
        for (int e = 0; e < 4; e++) Kbuf[r * DH_ + c0 + m * 4 + e] = f2bf(x[e]);
      }
      #pragma unroll
      for (int m = 0; m < 8; m++) {
        f4 x = ((const f4*)vp)[m];
        #pragma unroll
        for (int e = 0; e < 4; e++) VTb[(c0 + m * 4 + e) * VTS + r] = f2bf(x[e]);
      }
      if (t < BN) bvals[t] = bcol[gbase + j0 + t];
      if (jt + 1 < njt) {  // emits global_prefetch_b8 for next tile
        __builtin_prefetch(kp + (size_t)BN * E_, 0, 1);
        __builtin_prefetch(vp + (size_t)BN * E_, 0, 1);
      }
    }
    __syncthreads();

    // ---- S = Q K^T : hoist ALL 8 K-fragments, then chain 8 WMMAs ----
    // (batch the ds_load_b128s so WMMAs drain them with partial dscnt waits)
    Frag kf[8];
    #pragma unroll
    for (int n = 0; n < 2; n++)
      #pragma unroll
      for (int c = 0; c < 4; c++) {
        const unsigned short* kr = &Kbuf[(n * 16 + ln) * DH_ + c * 32 + g * 8];
        kf[n * 4 + c].h2[0] = *(const ushort8*)(kr);
        kf[n * 4 + c].h2[1] = *(const ushort8*)(kr + 16);
      }
    v8f sfr[2];
    #pragma unroll
    for (int n = 0; n < 2; n++) {
      v8f s = vz;
      #pragma unroll
      for (int c = 0; c < 4; c++)
        s = __builtin_amdgcn_wmma_f32_16x16x32_bf16(false, qa[c].v, false, kf[n * 4 + c].v,
                                                    (short)0, s, false, false);
      sfr[n] = s;
    }

    // ---- per-row online max/rescale:  m_new = max(m_old, a_i + max_{j<=i} b_j) ----
    if (lane < 16) {
      int i_l = irow0 + lane;
      float tmax = -__builtin_inff();
      #pragma unroll 8
      for (int j = 0; j < BN; j++)
        if (j0 + j <= i_l) tmax = fmaxf(tmax, bvals[j]);
      float mo = mrow[w][lane];
      float mn = fmaxf(mo, a_l + tmax);
      float scl = (mo == -__builtin_inff()) ? 0.f : __expf(mo - mn);
      srow[w][lane] *= scl;
      scm[w][lane] = scl;
      emm[w][lane] = a_l - mn;
      mrow[w][lane] = mn;
    }

    float sclM[8], emM[8];
    #pragma unroll
    for (int r = 0; r < 8; r++) { int M = r + 8 * g; sclM[r] = scm[w][M]; emM[r] = emm[w][M]; }
    #pragma unroll
    for (int d = 0; d < 8; d++)
      #pragma unroll
      for (int r = 0; r < 8; r++) acc[d][r] *= sclM[r];

    // ---- P = qk * exp(a_i + b_j - m_i) / sqrt(DH), causal mask, stage bf16 ----
    #pragma unroll
    for (int n = 0; n < 2; n++) {
      int j = j0 + n * 16 + ln;
      float bj = bvals[n * 16 + ln];
      #pragma unroll
      for (int r = 0; r < 8; r++) {
        int i = irow0 + r + 8 * g;
        float p = 0.f;
        if (j <= i) p = sfr[n][r] * invsqrt * __expf(emM[r] + bj);
        Pst[w][(r + 8 * g) * BN + n * 16 + ln] = f2bf(p);
      }
    }

    // ---- stabilizer row sums: srow += sum_j P (LDS in-order within wave) ----
    if (lane < 16) {
      const unsigned short* pr = &Pst[w][lane * BN];
      float sm = 0.f;
      #pragma unroll
      for (int j = 0; j < BN; j++) sm += bf2f(pr[j]);
      srow[w][lane] += sm;
    }

    // ---- O += P V : hoist P-fragment and ALL 8 V-fragments, then 8 WMMAs ----
    Frag pa;
    {
      const unsigned short* pr = &Pst[w][ln * BN + g * 8];
      pa.h2[0] = *(const ushort8*)(pr);
      pa.h2[1] = *(const ushort8*)(pr + 16);
    }
    Frag vbf[8];
    #pragma unroll
    for (int d = 0; d < 8; d++) {
      const unsigned short* vr = &VTb[(d * 16 + ln) * VTS + g * 8];
      vbf[d].h2[0] = *(const ushort8*)(vr);
      vbf[d].h2[1] = *(const ushort8*)(vr + 16);
    }
    #pragma unroll
    for (int d = 0; d < 8; d++)
      acc[d] = __builtin_amdgcn_wmma_f32_16x16x32_bf16(false, pa.v, false, vbf[d].v,
                                                       (short)0, acc[d], false, false);
    __syncthreads();
  }

  // ---- normalizer = max(|sum C|, exp(-m)) ; divide; stage rows ----
  if (lane < 16) {
    float nn = fmaxf(fabsf(srow[w][lane]), __expf(-mrow[w][lane]));
    scm[w][lane] = 1.f / (nn + 1e-6f);
  }
  float invM[8];
  #pragma unroll
  for (int r = 0; r < 8; r++) invM[r] = scm[w][r + 8 * g];
  #pragma unroll
  for (int d = 0; d < 8; d++)
    #pragma unroll
    for (int r = 0; r < 8; r++)
      outst[w][(r + 8 * g) * OST + d * 16 + ln] = acc[d][r] * invM[r];

  // ---- per-row multihead layernorm over DH, write (B,S,E) ----
  {
    int row = lane >> 1, half = lane & 1;  // 2 lanes per row
    const float* xr = &outst[w][row * OST + half * 64];
    float sm = 0.f, sq = 0.f;
    #pragma unroll 16
    for (int c = 0; c < 64; c++) { float x = xr[c]; sm += x; sq += x * x; }
    sm += __shfl_xor(sm, 1, 32);
    sq += __shfl_xor(sq, 1, 32);
    float mu = sm * (1.f / DH_);
    float var = sq * (1.f / DH_) - mu * mu;
    float rs = rsqrtf(var + 1e-6f);
    int i = irow0 + row;
    float* op = out + ((size_t)(b * S_ + i)) * E_ + h * DH_ + half * 64;
    const float* wp = lnw + h * DH_ + half * 64;
    #pragma unroll 16
    for (int c = 0; c < 64; c++) op[c] = (xr[c] - mu) * rs * wp[c];
  }
}

// ---------------------------------------------------------------------------
extern "C" void kernel_launch(void* const* d_in, const int* in_sizes, int n_in,
                              void* d_out, int out_size, void* d_ws, size_t ws_size,
                              hipStream_t stream) {
  (void)in_sizes; (void)n_in; (void)out_size; (void)ws_size;
  const float* q   = (const float*)d_in[0];
  const float* k   = (const float*)d_in[1];
  const float* v   = (const float*)d_in[2];
  const float* igk = (const float*)d_in[3];
  const float* igb = (const float*)d_in[4];
  const float* fgk = (const float*)d_in[5];
  const float* fgb = (const float*)d_in[6];
  const float* lnw = (const float*)d_in[7];
  float* out = (float*)d_out;

  float* ws = (float*)d_ws;
  const size_t NS = (size_t)B_ * NH_ * S_;
  float* igp = ws;
  float* fgp = ws + NS;
  float* cs  = ws + 2 * NS;
  float* bc  = ws + 3 * NS;

  hipLaunchKernelGGL(gates_kernel, dim3(B_ * S_), dim3(256), 0, stream,
                     q, k, v, igk, igb, fgk, fgb, igp, fgp);
  hipLaunchKernelGGL(scan_kernel, dim3(B_ * NH_), dim3(256), 0, stream,
                     igp, fgp, cs, bc);
  hipLaunchKernelGGL(mlstm_attn_kernel, dim3(B_ * NH_ * (S_ / BM)), dim3(128), 0, stream,
                     q, k, v, cs, bc, lnw, out);
}
